// LaplacianLinearAttention_60559038873884
// MI455X (gfx1250) — compile-verified, matrix operand order checked by reading, not run on hardware
//
#include <hip/hip_runtime.h>

// ---------------------------------------------------------------------------
// LaplacianLinearAttention for MI455X (gfx1250, wave32, WMMA).
// f32 storage everywhere; f16 WMMA (f32 accum) for GEMM stages;
// f32 VALU for L1-kernel / exp / Newton-Schulz / conv / stats.
// CDNA5 paths: v_wmma_f32_16x16x32_f16, global_load_async_to_lds_b128
// (ASYNCcnt), global_prefetch_b8.
// ---------------------------------------------------------------------------

typedef __attribute__((ext_vector_type(16))) _Float16 v16h;
typedef __attribute__((ext_vector_type(8)))  float    v8f;
typedef __attribute__((ext_vector_type(4)))  int      v4i;

typedef __attribute__((address_space(1))) v4i* g_v4i_p;   // global
typedef __attribute__((address_space(3))) v4i* l_v4i_p;   // LDS

#if defined(__HIP_DEVICE_COMPILE__) && __has_builtin(__builtin_amdgcn_global_load_async_to_lds_b128)
#define HAVE_ASYNC_LDS 1
#else
#define HAVE_ASYNC_LDS 0
#endif

#if defined(__HIP_DEVICE_COMPILE__) && __has_builtin(__builtin_amdgcn_s_wait_asynccnt)
#define ASYNC_WAIT() __builtin_amdgcn_s_wait_asynccnt(0)
#elif defined(__HIP_DEVICE_COMPILE__)
#define ASYNC_WAIT() asm volatile("s_wait_asynccnt 0x0" ::: "memory")
#else
#define ASYNC_WAIT()
#endif

constexpr int NB = 2;      // batch
constexpr int NN = 1024;   // tokens
constexpr int NC = 256;    // channels
constexpr int NH = 8;      // heads
constexpr int HD = 32;     // head dim
constexpr int MM = 64;     // landmarks (8x8 after 4x4 pool of 32x32)

constexpr size_t Q_SZ  = (size_t)NB * NH * NN * HD;  // 524288
constexpr size_t VL_SZ = (size_t)NB * NN * NC;       // 524288
constexpr size_t CM_SZ = (size_t)NB * NH * NN * MM;  // 1048576
constexpr size_t LM_SZ = (size_t)NB * NH * MM * HD;  // 32768
constexpr size_t W_SZ  = (size_t)NB * NH * MM * MM;  // 65536
constexpr size_t CT_SZ = (size_t)NB * NH * MM * HD;  // 32768
constexpr size_t ST_SZ = (size_t)NH * MM;            // 512

constexpr size_t OFF_Q   = 0;
constexpr size_t OFF_K   = OFF_Q   + Q_SZ;
constexpr size_t OFF_V   = OFF_K   + Q_SZ;
constexpr size_t OFF_VL  = OFF_V   + Q_SZ;
constexpr size_t OFF_AT  = OFF_VL  + VL_SZ;
constexpr size_t OFF_CQ  = OFF_AT  + VL_SZ;
constexpr size_t OFF_CK  = OFF_CQ  + CM_SZ;
constexpr size_t OFF_QLM = OFF_CK  + CM_SZ;
constexpr size_t OFF_KLM = OFF_QLM + LM_SZ;
constexpr size_t OFF_W   = OFF_KLM + LM_SZ;
constexpr size_t OFF_WI  = OFF_W   + W_SZ;
constexpr size_t OFF_CTX = OFF_WI  + W_SZ;
constexpr size_t OFF_CT2 = OFF_CTX + CT_SZ;
constexpr size_t OFF_MU  = OFF_CT2 + CT_SZ;
constexpr size_t OFF_RI  = OFF_MU  + ST_SZ;

// f32 -> f16 fragment load; p points at (row_or_col, k0); K-contiguous source.
__device__ __forceinline__ v16h cvt_frag(const float* p, int half) {
  v16h f;
#pragma unroll
  for (int i = 0; i < 8; ++i) {
    f[i]     = (_Float16)p[half * 8 + i];
    f[i + 8] = (_Float16)p[16 + half * 8 + i];
  }
  return f;
}

// ---------------------------------------------------------------------------
// 1) QKV GEMM: out[r, col] = sum_k x[r,k] * qkv_w[col,k] + qkv_b[col]
//    scattered to q/k/v buffers in (B,nh,N,hd) layout.
// grid (2048/16, 768/(8*16)) = (128, 6), block 256 (8 waves, 16x16 tile each)
// ---------------------------------------------------------------------------
__global__ void k_qkv(const float* __restrict__ x, const float* __restrict__ w,
                      const float* __restrict__ bias, float* __restrict__ q,
                      float* __restrict__ k, float* __restrict__ v) {
  const int t = threadIdx.x, wave = t >> 5, lane = t & 31;
  const int row = lane & 15, half = lane >> 4;
  const int r0 = blockIdx.x * 16;
  const int c0 = (blockIdx.y * 8 + wave) * 16;
  const float* arow = x + (size_t)(r0 + row) * NC;
  const float* brow = w + (size_t)(c0 + row) * NC;
  v8f acc = {};
  for (int k0 = 0; k0 < NC; k0 += 32) {
    __builtin_prefetch(arow + k0 + 32, 0, 0);  // speculative next-tile prefetch
    __builtin_prefetch(brow + k0 + 32, 0, 0);
    v16h a  = cvt_frag(arow + k0, half);
    v16h bb = cvt_frag(brow + k0, half);
    acc = __builtin_amdgcn_wmma_f32_16x16x32_f16(false, a, false, bb,
                                                 (short)0, acc, false, false);
  }
  const int col = c0 + row;
  const int typ = col >> 8;       // 0=q 1=k 2=v
  const int rem = col & 255;
  const int h = rem >> 5, d = rem & 31;
  float* dst = (typ == 0) ? q : ((typ == 1) ? k : v);
  const float bv = bias[col];
#pragma unroll
  for (int j = 0; j < 8; ++j) {
    const int r = r0 + j + 8 * half;
    const int b = r >> 10, n = r & 1023;
    dst[(((size_t)b * NH + h) * NN + n) * HD + d] = acc[j] + bv;
  }
}

// ---------------------------------------------------------------------------
// 2) Depthwise 3x3 conv (SAME) on v through the torch-faithful reshape.
//    v_spatial[b,ch,h,w] = v_bnc flat index ch*1024+h*32+w, where
//    v_bnc[b,n,c] = v[b, c/32, n, c%32].  Output to vlocal (B,N,C).
// ---------------------------------------------------------------------------
__global__ void k_dwc(const float* __restrict__ v, const float* __restrict__ w,
                      const float* __restrict__ bias, float* __restrict__ vlocal) {
  const int idx = blockIdx.x * blockDim.x + threadIdx.x;  // < 2*256*1024
  const int b = idx >> 18;
  const int rem = idx & 262143;
  const int ch = rem >> 10;
  const int pix = rem & 1023;
  const int hh = pix >> 5, ww = pix & 31;
  float acc = bias[ch];
#pragma unroll
  for (int dy = -1; dy <= 1; ++dy)
#pragma unroll
    for (int dx = -1; dx <= 1; ++dx) {
      const int y = hh + dy, xcol = ww + dx;
      if (y >= 0 && y < 32 && xcol >= 0 && xcol < 32) {
        const int flat = ch * 1024 + y * 32 + xcol;
        const int np = flat >> 8, cp = flat & 255;
        const float val =
            v[(((size_t)b * NH + (cp >> 5)) * NN + np) * HD + (cp & 31)];
        acc += val * w[ch * 9 + (dy + 1) * 3 + (dx + 1)];
      }
    }
  vlocal[((size_t)b * NN + pix) * NC + ch] = acc;
}

// ---------------------------------------------------------------------------
// 3) 4x4 avg-pool of q/k spatial + to_landmark -> q_lm/k_lm (B,nh,M,hd)
// ---------------------------------------------------------------------------
__global__ void k_landmarks(const float* __restrict__ q, const float* __restrict__ k,
                            float* __restrict__ qlm, float* __restrict__ klm) {
  const int idx = blockIdx.x * blockDim.x + threadIdx.x;  // < 32768
  const int d = idx & 31;
  const int m = (idx >> 5) & 63;
  const int h = (idx >> 11) & 7;
  const int b = idx >> 14;
  const int ch = h * HD + d;
  const int i0 = (m >> 3) * 4, j0 = (m & 7) * 4;
  float sq = 0.f, sk = 0.f;
  for (int ii = 0; ii < 4; ++ii)
    for (int jj = 0; jj < 4; ++jj) {
      const int flat = ch * 1024 + (i0 + ii) * 32 + (j0 + jj);
      const int np = flat >> 8, cp = flat & 255;
      const size_t o = (((size_t)b * NH + (cp >> 5)) * NN + np) * HD + (cp & 31);
      sq += q[o];
      sk += k[o];
    }
  qlm[idx] = sq * (1.f / 16.f);
  klm[idx] = sk * (1.f / 16.f);
}

// ---------------------------------------------------------------------------
// 4) W = exp(-||q_lm - k_lm||_1 / 4)  (B,nh,64,64); one block per (b,h)
// ---------------------------------------------------------------------------
__global__ void k_wmat(const float* __restrict__ qlm, const float* __restrict__ klm,
                       float* __restrict__ Wm) {
  __shared__ float Qs[MM * HD], Ks[MM * HD];
  const int t = threadIdx.x, bh = blockIdx.x;
  const float* qs = qlm + (size_t)bh * MM * HD;
  const float* ks = klm + (size_t)bh * MM * HD;
  for (int i = 0; i < 8; ++i) { Qs[t * 8 + i] = qs[t * 8 + i]; Ks[t * 8 + i] = ks[t * 8 + i]; }
  __syncthreads();
  float* dst = Wm + (size_t)bh * MM * MM;
  for (int i = 0; i < 16; ++i) {
    const int e = t * 16 + i;
    const int r = e >> 6, c = e & 63;
    float l1 = 0.f;
    for (int dd = 0; dd < HD; ++dd) l1 += fabsf(Qs[r * HD + dd] - Ks[c * HD + dd]);
    dst[e] = __expf(-l1 * 0.25f);
  }
}

// ---------------------------------------------------------------------------
// 5) Newton-Schulz pseudo-inverse of W (+eps I), 5 iters, f32 in LDS.
//    One block per (b,h); 256 threads x 16 elems.
// ---------------------------------------------------------------------------
__global__ void k_ns(const float* __restrict__ Wm, float* __restrict__ Winv) {
  __shared__ float Ws[MM * MM];
  __shared__ float Xs[MM * MM];
  __shared__ float Ts[MM * MM];
  __shared__ float red[256];
  const int t = threadIdx.x, bh = blockIdx.x;
  const float* src = Wm + (size_t)bh * MM * MM;
  float ss = 0.f;
  for (int i = 0; i < 16; ++i) {
    const int e = t * 16 + i;
    float w = src[e];
    if ((e >> 6) == (e & 63)) w += 1e-4f;  // NS_EPS * I
    Ws[e] = w;
    ss += w * w;
  }
  red[t] = ss;
  __syncthreads();
  for (int s = 128; s > 0; s >>= 1) {
    if (t < s) red[t] += red[t + s];
    __syncthreads();
  }
  const float alpha = 2.0f / (sqrtf(red[0]) + 1e-8f);
  for (int i = 0; i < 16; ++i) {
    const int e = t * 16 + i;
    const int r = e >> 6, c = e & 63;
    Xs[e] = alpha * Ws[c * MM + r];  // alpha * W_eps^T
  }
  __syncthreads();
  for (int it = 0; it < 5; ++it) {
    for (int i = 0; i < 16; ++i) {          // T = W_eps @ X
      const int e = t * 16 + i;
      const int r = e >> 6, c = e & 63;
      float s2 = 0.f;
      for (int kk = 0; kk < MM; ++kk) s2 += Ws[r * MM + kk] * Xs[kk * MM + c];
      Ts[e] = s2;
    }
    __syncthreads();
    float tmp[16];
    for (int i = 0; i < 16; ++i) {          // X = 2X - X@T
      const int e = t * 16 + i;
      const int r = e >> 6, c = e & 63;
      float s2 = 0.f;
      for (int kk = 0; kk < MM; ++kk) s2 += Xs[r * MM + kk] * Ts[kk * MM + c];
      tmp[i] = 2.f * Xs[e] - s2;
    }
    __syncthreads();
    for (int i = 0; i < 16; ++i) Xs[t * 16 + i] = tmp[i];
    __syncthreads();
  }
  float* dst = Winv + (size_t)bh * MM * MM;
  for (int i = 0; i < 16; ++i) dst[t * 16 + i] = Xs[t * 16 + i];
}

// ---------------------------------------------------------------------------
// 6) Laplacian kernel C[n,m] = exp(-||src_n - lm_m||_1 / 4).
//    One block per (b,h,64-row chunk); LDS-staged rows + landmarks.
// ---------------------------------------------------------------------------
__global__ void k_lap(const float* __restrict__ src, const float* __restrict__ lm,
                      float* __restrict__ dst) {
  __shared__ float Ss[64 * HD], Ls[MM * HD];
  const int t = threadIdx.x, bid = blockIdx.x;
  const int nblk = bid & 15, h = (bid >> 4) & 7, b = bid >> 7;
  const int bh = b * NH + h;
  const int n0 = nblk * 64;
  const float* sp = src + ((size_t)bh * NN + n0) * HD;
  const float* lp = lm + (size_t)bh * MM * HD;
  for (int i = 0; i < 8; ++i) { Ss[t * 8 + i] = sp[t * 8 + i]; Ls[t * 8 + i] = lp[t * 8 + i]; }
  __syncthreads();
  float* dp = dst + ((size_t)bh * NN + n0) * MM;
  for (int i = 0; i < 16; ++i) {
    const int e = t * 16 + i;
    const int nl = e >> 6, m = e & 63;
    float l1 = 0.f;
    for (int dd = 0; dd < HD; ++dd) l1 += fabsf(Ss[nl * HD + dd] - Ls[m * HD + dd]);
    dp[e] = __expf(-l1 * 0.25f);
  }
}

// ---------------------------------------------------------------------------
// 7) Whitening stats of C_q over (b,n) per (h,m): mu, 1/sqrt(var_unbiased+eps)
// ---------------------------------------------------------------------------
__global__ void k_stats(const float* __restrict__ cq, float* __restrict__ mu,
                        float* __restrict__ rinv) {
  __shared__ float s1[256], s2[256];
  const int t = threadIdx.x;
  const int hm = blockIdx.x;  // h*64+m
  const int h = hm >> 6, m = hm & 63;
  float a = 0.f, b2 = 0.f;
  for (int i = 0; i < 8; ++i) {
    const int sidx = t + 256 * i;  // 0..2047 over (b,n)
    const int b = sidx >> 10, n = sidx & 1023;
    const float v = cq[(((size_t)b * NH + h) * NN + n) * MM + m];
    a += v;
    b2 += v * v;
  }
  s1[t] = a; s2[t] = b2;
  __syncthreads();
  for (int s = 128; s > 0; s >>= 1) {
    if (t < s) { s1[t] += s1[t + s]; s2[t] += s2[t + s]; }
    __syncthreads();
  }
  if (t == 0) {
    const float mean = s1[0] / 2048.f;
    const float var = (s2[0] - 2048.f * mean * mean) / 2047.f;  // ddof=1
    mu[hm] = mean;
    rinv[hm] = rsqrtf(fmaxf(var, 0.f) + 1e-5f);
  }
}

// ---------------------------------------------------------------------------
// 8) context[m,d] = sum_n C_k[n,m] * v[n,d]  (per b,h), WMMA.
//    A = C_k^T via manual f16 LDS transpose; B = v chunk staged raw (f32)
//    into LDS via GLOBAL_LOAD_ASYNC_TO_LDS_B128 (ASYNCcnt) when available.
//    8 waves: 4 m-tiles x 2 d-tiles; K=1024 in 64-chunks.
// ---------------------------------------------------------------------------
__global__ void k_context(const float* __restrict__ ck, const float* __restrict__ v,
                          float* __restrict__ ctx) {
  __shared__ _Float16 As[MM * MM];                 // [m][n_local] (transposed)
  __shared__ __align__(16) float Bsf[MM * HD];     // [n_local][d], raw f32 copy
  const int t = threadIdx.x, bh = blockIdx.x;
  const int wave = t >> 5, lane = t & 31;
  const int row = lane & 15, half = lane >> 4;
  const int mt = wave >> 1, dt = wave & 1;
  const float* ckb = ck + (size_t)bh * NN * MM;
  const float* vb = v + (size_t)bh * NN * HD;
  v8f acc = {};
  for (int n0 = 0; n0 < NN; n0 += 64) {
    __syncthreads();
#if HAVE_ASYNC_LDS
    // 64 rows x 32 f32 = 512 x 16B; 2 async b128 copies per thread.
#pragma unroll
    for (int i = 0; i < 2; ++i) {
      const int e = t * 2 + i;            // 0..511
      const int nl = e >> 3, seg = e & 7; // row, 16B segment
      __builtin_amdgcn_global_load_async_to_lds_b128(
          (g_v4i_p)(vb + (size_t)(n0 + nl) * HD + seg * 4),
          (l_v4i_p)&Bsf[nl * HD + seg * 4],
          0, 0);
    }
#else
    for (int i = 0; i < 8; ++i) {
      const int e = t * 8 + i;
      Bsf[e] = vb[(size_t)(n0 + (e >> 5)) * HD + (e & 31)];
    }
#endif
    {
      const int nl = t >> 2;
      const int mbase = (t & 3) * 16;
      const float* srcp = ckb + (size_t)(n0 + nl) * MM + mbase;
      for (int i = 0; i < 16; ++i) As[(mbase + i) * MM + nl] = (_Float16)srcp[i];
    }
#if HAVE_ASYNC_LDS
    ASYNC_WAIT();
#endif
    __syncthreads();
    for (int k0 = 0; k0 < 64; k0 += 32) {
      v16h a, bf;
#pragma unroll
      for (int i = 0; i < 8; ++i) {
        a[i]      = As[(mt * 16 + row) * MM + k0 + half * 8 + i];
        a[i + 8]  = As[(mt * 16 + row) * MM + k0 + 16 + half * 8 + i];
        bf[i]     = (_Float16)Bsf[(k0 + half * 8 + i) * HD + dt * 16 + row];
        bf[i + 8] = (_Float16)Bsf[(k0 + 16 + half * 8 + i) * HD + dt * 16 + row];
      }
      acc = __builtin_amdgcn_wmma_f32_16x16x32_f16(false, a, false, bf,
                                                   (short)0, acc, false, false);
    }
  }
  float* dst = ctx + (size_t)bh * MM * HD;
#pragma unroll
  for (int j = 0; j < 8; ++j) {
    const int m = mt * 16 + j + 8 * half;
    dst[m * HD + dt * 16 + row] = acc[j];
  }
}

// ---------------------------------------------------------------------------
// 9) ctx2 = W_inv @ ctx  (64x64 @ 64x32 per b,h), f32 in LDS
// ---------------------------------------------------------------------------
__global__ void k_ctx2(const float* __restrict__ winv, const float* __restrict__ ctx,
                       float* __restrict__ ctx2) {
  __shared__ float Ws[MM * MM];
  __shared__ float Cs[MM * HD];
  const int t = threadIdx.x, bh = blockIdx.x;
  const float* wsrc = winv + (size_t)bh * MM * MM;
  const float* csrc = ctx + (size_t)bh * MM * HD;
  for (int i = 0; i < 16; ++i) Ws[t * 16 + i] = wsrc[t * 16 + i];
  for (int i = 0; i < 8; ++i) Cs[t * 8 + i] = csrc[t * 8 + i];
  __syncthreads();
  float* dst = ctx2 + (size_t)bh * MM * HD;
  for (int i = 0; i < 8; ++i) {
    const int e = t * 8 + i;
    const int m = e >> 5, d = e & 31;
    float s = 0.f;
    for (int kk = 0; kk < MM; ++kk) s += Ws[m * MM + kk] * Cs[kk * HD + d];
    dst[e] = s;
  }
}

// ---------------------------------------------------------------------------
// 10) global_attn = whiten(C_q) @ ctx2, fused +v_local, to attn (B,N,C)
//     One block per (b,h,64-row chunk); 8 waves: 4 n-tiles x 2 d-tiles; K=64.
// ---------------------------------------------------------------------------
__global__ void k_attn(const float* __restrict__ cq, const float* __restrict__ mu,
                       const float* __restrict__ rinv, const float* __restrict__ ctx2,
                       const float* __restrict__ vlocal, float* __restrict__ attn) {
  __shared__ float C2s[MM * HD];
  __shared__ float Mus[MM], Ris[MM];
  const int t = threadIdx.x, bid = blockIdx.x;
  const int nblk = bid & 15, h = (bid >> 4) & 7, b = bid >> 7;
  const int bh = b * NH + h;
  for (int i = 0; i < 8; ++i) C2s[t * 8 + i] = ctx2[(size_t)bh * MM * HD + t * 8 + i];
  if (t < MM) { Mus[t] = mu[h * MM + t]; Ris[t] = rinv[h * MM + t]; }
  __syncthreads();
  const int wave = t >> 5, lane = t & 31;
  const int row = lane & 15, half = lane >> 4;
  const int nt = wave >> 1, dt = wave & 1;
  const int n0 = nblk * 64;
  const float* cqrow = cq + ((size_t)bh * NN + n0 + nt * 16 + row) * MM;
  v8f acc = {};
  for (int k0 = 0; k0 < MM; k0 += 32) {
    v16h a, bf;
#pragma unroll
    for (int i = 0; i < 8; ++i) {
      const int ka = k0 + half * 8 + i;
      const int kb = k0 + 16 + half * 8 + i;
      a[i]      = (_Float16)((cqrow[ka] - Mus[ka]) * Ris[ka]);
      a[i + 8]  = (_Float16)((cqrow[kb] - Mus[kb]) * Ris[kb]);
      bf[i]     = (_Float16)C2s[ka * HD + dt * 16 + row];
      bf[i + 8] = (_Float16)C2s[kb * HD + dt * 16 + row];
    }
    acc = __builtin_amdgcn_wmma_f32_16x16x32_f16(false, a, false, bf,
                                                 (short)0, acc, false, false);
  }
  const int d = dt * 16 + row;
  const int c = h * HD + d;
#pragma unroll
  for (int j = 0; j < 8; ++j) {
    const int n = n0 + nt * 16 + j + 8 * half;
    const size_t o = ((size_t)b * NN + n) * NC + c;
    attn[o] = acc[j] + vlocal[o];
  }
}

// ---------------------------------------------------------------------------
// 11) Output projection: out = attn @ proj_w^T + proj_b  (2048x256 @ 256x256)
// ---------------------------------------------------------------------------
__global__ void k_proj(const float* __restrict__ a_in, const float* __restrict__ w,
                       const float* __restrict__ bias, float* __restrict__ out) {
  const int t = threadIdx.x, wave = t >> 5, lane = t & 31;
  const int row = lane & 15, half = lane >> 4;
  const int r0 = blockIdx.x * 16;
  const int c0 = (blockIdx.y * 8 + wave) * 16;
  const float* arow = a_in + (size_t)(r0 + row) * NC;
  const float* brow = w + (size_t)(c0 + row) * NC;
  v8f acc = {};
  for (int k0 = 0; k0 < NC; k0 += 32) {
    __builtin_prefetch(arow + k0 + 32, 0, 0);
    __builtin_prefetch(brow + k0 + 32, 0, 0);
    v16h a  = cvt_frag(arow + k0, half);
    v16h bb = cvt_frag(brow + k0, half);
    acc = __builtin_amdgcn_wmma_f32_16x16x32_f16(false, a, false, bb,
                                                 (short)0, acc, false, false);
  }
  const int col = c0 + row;
  const float bv = bias[col];
#pragma unroll
  for (int j = 0; j < 8; ++j) {
    out[(size_t)(r0 + j + 8 * half) * NC + col] = acc[j] + bv;
  }
}

// ---------------------------------------------------------------------------
extern "C" void kernel_launch(void* const* d_in, const int* in_sizes, int n_in,
                              void* d_out, int out_size, void* d_ws, size_t ws_size,
                              hipStream_t stream) {
  (void)in_sizes; (void)n_in; (void)out_size; (void)ws_size;
  const float* x      = (const float*)d_in[0];
  const float* qkv_w  = (const float*)d_in[1];
  const float* qkv_b  = (const float*)d_in[2];
  const float* proj_w = (const float*)d_in[3];
  const float* proj_b = (const float*)d_in[4];
  const float* dwc_w  = (const float*)d_in[5];
  const float* dwc_b  = (const float*)d_in[6];

  float* ws = (float*)d_ws;
  float* q    = ws + OFF_Q;
  float* k    = ws + OFF_K;
  float* v    = ws + OFF_V;
  float* vloc = ws + OFF_VL;
  float* attn = ws + OFF_AT;
  float* cq   = ws + OFF_CQ;
  float* ckb  = ws + OFF_CK;
  float* qlm  = ws + OFF_QLM;
  float* klm  = ws + OFF_KLM;
  float* Wm   = ws + OFF_W;
  float* Winv = ws + OFF_WI;
  float* ctx  = ws + OFF_CTX;
  float* ctx2 = ws + OFF_CT2;
  float* mu   = ws + OFF_MU;
  float* rinv = ws + OFF_RI;
  float* out  = (float*)d_out;

  k_qkv<<<dim3(128, 6), 256, 0, stream>>>(x, qkv_w, qkv_b, q, k, v);
  k_dwc<<<2048, 256, 0, stream>>>(v, dwc_w, dwc_b, vloc);
  k_landmarks<<<128, 256, 0, stream>>>(q, k, qlm, klm);
  k_wmat<<<16, 256, 0, stream>>>(qlm, klm, Wm);
  k_ns<<<16, 256, 0, stream>>>(Wm, Winv);
  k_lap<<<256, 256, 0, stream>>>(q, klm, cq);   // C_q = lap(q, k_lm)
  k_lap<<<256, 256, 0, stream>>>(k, qlm, ckb);  // C_k = lap(k, q_lm)
  k_stats<<<512, 256, 0, stream>>>(cq, mu, rinv);
  k_context<<<16, 256, 0, stream>>>(ckb, v, ctx);
  k_ctx2<<<16, 256, 0, stream>>>(Winv, ctx, ctx2);
  k_attn<<<256, 256, 0, stream>>>(cq, mu, rinv, ctx2, vloc, attn);
  k_proj<<<dim3(128, 2), 256, 0, stream>>>(attn, proj_w, proj_b, out);
}
